// RecNN_1331439862191
// MI455X (gfx1250) — matrix operand order repaired and checked by reading, not compiled, not run on hardware
//
#include <hip/hip_runtime.h>

// ---------------------------------------------------------------------------
// RecNN tree reduction for MI455X (gfx1250, wave32, WMMA).
// All 1024x1024 GEMMs run on v_wmma_f32_16x16x32_bf16 with fp32 accumulation,
// using a hi/lo bf16 split of every operand ("bf16x3" emulated fp32):
//     A*B ~= Ah*Bh + Ah*Bl + Al*Bh            (error ~2^-16)
// Weight tiles are staged in LDS via the CDNA5 async global->LDS path
// (GLOBAL_LOAD_ASYNC_TO_LDS_B128 + s_wait_asynccnt), triple-buffered and
// shared by all 8 waves of a block (8x L2 traffic cut). Each wave computes a
// 32(M) x 64(N) tile (8 WMMA accumulators), 24 WMMAs per K=32 step.
// ---------------------------------------------------------------------------

typedef __attribute__((ext_vector_type(16))) __bf16 v16bf;
typedef __attribute__((ext_vector_type(8)))  __bf16 v8bf;
typedef __attribute__((ext_vector_type(8)))  float  v8f;
typedef __attribute__((ext_vector_type(4)))  float  v4f;
typedef int v4i_gcc __attribute__((vector_size(16)));   // async-LDS builtin pointee

#define DHID 1024   // D_IN == D_H == D_OUT == 1024
#define NB   8      // batch
#define NL   1024   // sequence length
#define LPIT 40     // LDS row pitch in bf16 (80B: bank-staggered, 16B aligned)
#define NSTEP (DHID / 32)

#if __has_builtin(__builtin_amdgcn_global_load_async_to_lds_b128)
#define HAVE_ASYNC_LDS 1
#else
#define HAVE_ASYNC_LDS 0
#endif

// Issue a 32B global->LDS fragment copy (per thread).
__device__ __forceinline__ void fill32(const __bf16* g, __bf16* l) {
#if HAVE_ASYNC_LDS
    __builtin_amdgcn_global_load_async_to_lds_b128(
        (__attribute__((address_space(1))) v4i_gcc*)g,
        (__attribute__((address_space(3))) v4i_gcc*)l, 0, 0);
    __builtin_amdgcn_global_load_async_to_lds_b128(
        (__attribute__((address_space(1))) v4i_gcc*)(g + 8),
        (__attribute__((address_space(3))) v4i_gcc*)(l + 8), 0, 0);
#else
    union { v16bf v; v8bf h[2]; } t;
    t.v = *(const v16bf*)g;
    *(v8bf*)l = t.h[0];
    *(v8bf*)(l + 8) = t.h[1];
#endif
}

template <int N>
__device__ __forceinline__ void wait_async() {
#if HAVE_ASYNC_LDS
#if __has_builtin(__builtin_amdgcn_s_wait_asynccnt)
    __builtin_amdgcn_s_wait_asynccnt((unsigned short)N);
#else
    asm volatile("s_wait_asynccnt %0" :: "i"(N) : "memory");
#endif
#endif
}

// ---------------------------------------------------------------------------
// Split an f32 matrix into hi/lo bf16 (elementwise, layout-preserving).
// ---------------------------------------------------------------------------
__global__ void wsplit_kernel(const float* __restrict__ w,
                              __bf16* __restrict__ hi,
                              __bf16* __restrict__ lo, int n) {
    int i = blockIdx.x * blockDim.x + threadIdx.x;
    if (i < n) {
        float x = w[i];
        __bf16 h = (__bf16)x;
        hi[i] = h;
        lo[i] = (__bf16)(x - (float)h);
    }
}

// ---------------------------------------------------------------------------
// Fused GEMM + bias + ReLU, output split to hi/lo bf16.
//   out[r, :] = relu( A[r*rowStride + 0] @ W1^T + b1
//                   (+ A[r*rowStride + 1] @ W2^T + b2  if TWO) )
// AF32: A is raw f32 (leaves), split in-register; else A is hi/lo bf16.
// Block = 256 threads = 8 waves. Block tile: 256(M) x 64(N).
// Wave tile: 32(M) x 64(N) -> acc[2][4] 16x16 f32 tiles.
// B (weights) staged in LDS: 64 cols x 32 K, hi+lo, triple buffered (30KB),
// filled with async global->LDS DMA, one s_wait_asynccnt + barrier per step.
// ---------------------------------------------------------------------------
template <bool AF32, bool TWO>
__global__ void __launch_bounds__(256) tree_gemm(
    const float*  __restrict__ Af,
    const __bf16* __restrict__ Ahi, const __bf16* __restrict__ Alo,
    const __bf16* __restrict__ W1hi, const __bf16* __restrict__ W1lo,
    const float*  __restrict__ b1,
    const __bf16* __restrict__ W2hi, const __bf16* __restrict__ W2lo,
    const float*  __restrict__ b2,
    __bf16* __restrict__ Chi, __bf16* __restrict__ Clo,
    int M, int rowStride)
{
    __shared__ __bf16 ldsB[3][2][64 * LPIT];   // [buf][hi/lo][col*LPIT + k]

    const int tid  = threadIdx.x;
    const int wave = tid >> 5;
    const int lane = tid & 31;
    const int hl   = lane >> 4;          // half-wave: 0 or 1
    const int l15  = lane & 15;
    const int m0   = (blockIdx.y * 8 + wave) * 32;
    const int n0   = blockIdx.x * 64;

    // LDS-fill role of this thread: 128 threads per matrix (hi/lo),
    // each thread owns one (col, 16-wide K half) 32B fragment per step.
    const int f_mat  = tid >> 7;         // 0 = hi, 1 = lo
    const int f_col  = (tid >> 1) & 63;  // 0..63
    const int f_half = tid & 1;          // 0..1
    __bf16* lslot[3];
#pragma unroll
    for (int b = 0; b < 3; ++b)
        lslot[b] = &ldsB[b][f_mat][f_col * LPIT + f_half * 16];

    v8f acc[2][4] = {};                  // [Msub][Nsub] 16x16 f32 tiles

    // Clamp M rows (last tree level has M=8 < 16); stores are masked below.
    int mcl[2];
#pragma unroll
    for (int ms = 0; ms < 2; ++ms) mcl[ms] = min(m0 + ms * 16 + l15, M - 1);

    union AU { v16bf v; v8bf h[2]; };

    for (int pass = 0; pass < (TWO ? 2 : 1); ++pass) {
        const __bf16* Whi = (TWO && pass) ? W2hi : W1hi;
        const __bf16* Wlo = (TWO && pass) ? W2lo : W1lo;
        const __bf16* wfill =
            (f_mat ? Wlo : Whi) + (n0 + f_col) * DHID + f_half * 16;

        int aRowB[2];
#pragma unroll
        for (int ms = 0; ms < 2; ++ms)
            aRowB[ms] = mcl[ms] * rowStride + pass;

        // ---- Prologue: launch fills for K-steps 0 and 1.
        fill32(wfill, lslot[0]);
        fill32(wfill + 32, lslot[1]);
        wait_async<2>();                 // own step-0 fill complete
        __syncthreads();                 // everyone's step-0 fill complete

        for (int s = 0; s < NSTEP; ++s) {
            const int k0  = s * 32;
            const int buf = s % 3;

            // ---- A operands (16x32 bf16 per Msub): lane = row(l15),
            //      lanes 0-15:  K {k0..k0+7, k0+16..k0+23}
            //      lanes 16-31: K {k0+8..k0+15, k0+24..k0+31}
            AU ah[2], al[2];
            const int ka = k0 + hl * 8;
#pragma unroll
            for (int ms = 0; ms < 2; ++ms) {
                if (AF32) {
                    const float* ap = Af + aRowB[ms] * DHID + ka;
                    v4f fb[4];
                    fb[0] = *(const v4f*)(ap);
                    fb[1] = *(const v4f*)(ap + 4);
                    fb[2] = *(const v4f*)(ap + 16);
                    fb[3] = *(const v4f*)(ap + 20);
#pragma unroll
                    for (int i = 0; i < 16; ++i) {
                        float x = fb[i >> 2][i & 3];
                        __bf16 h = (__bf16)x;
                        ah[ms].v[i] = h;
                        al[ms].v[i] = (__bf16)(x - (float)h);
                    }
                } else {
                    const __bf16* aph = Ahi + aRowB[ms] * DHID + ka;
                    const __bf16* apl = Alo + aRowB[ms] * DHID + ka;
                    ah[ms].h[0] = *(const v8bf*)(aph);
                    ah[ms].h[1] = *(const v8bf*)(aph + 16);
                    al[ms].h[0] = *(const v8bf*)(apl);
                    al[ms].h[1] = *(const v8bf*)(apl + 16);
                }
            }

            // ---- B operands from LDS (32x16 bf16): lane = column(l15),
            //      lanes 0-15 -> K {k0..k0+7, k0+16..k0+23}, lanes 16-31 -> K {k0+8..k0+15, k0+24..k0+31}.
#pragma unroll
            for (int sub = 0; sub < 4; ++sub) {
                const int boff = (sub * 16 + l15) * LPIT + hl * 8;
                const __bf16* bp = &ldsB[buf][0][boff];
                const __bf16* bq = &ldsB[buf][1][boff];
                AU bh, bl;
                bh.h[0] = *(const v8bf*)bp;
                bh.h[1] = *(const v8bf*)(bp + 16);
                bl.h[0] = *(const v8bf*)bq;
                bl.h[1] = *(const v8bf*)(bq + 16);
#pragma unroll
                for (int ms = 0; ms < 2; ++ms) {
                    acc[ms][sub] = __builtin_amdgcn_wmma_f32_16x16x32_bf16(
                        false, ah[ms].v, false, bh.v, (short)0, acc[ms][sub],
                        false, false);
                    acc[ms][sub] = __builtin_amdgcn_wmma_f32_16x16x32_bf16(
                        false, ah[ms].v, false, bl.v, (short)0, acc[ms][sub],
                        false, false);
                    acc[ms][sub] = __builtin_amdgcn_wmma_f32_16x16x32_bf16(
                        false, al[ms].v, false, bh.v, (short)0, acc[ms][sub],
                        false, false);
                }
            }

            // Refill the buffer last read in step s-1 with K-step s+2;
            // wait for own s+1 fill, then barrier => everyone's s+1 fill is
            // visible and nobody still reads the buffer being refilled.
            if (s + 2 < NSTEP) {
                fill32(wfill + (s + 2) * 32, lslot[(s + 2) % 3]);
                wait_async<2>();
            } else {
                wait_async<0>();
            }
            __syncthreads();
        }
    }

    // ---- Epilogue: bias + ReLU + hi/lo split store.
    // C/D layout: VGPR g -> M = m0 + ms*16 + hl*8 + g, N = n0 + sub*16 + l15.
#pragma unroll
    for (int sub = 0; sub < 4; ++sub) {
        const int col = n0 + sub * 16 + l15;
        float bias = b1[col];
        if (TWO) bias += b2[col];
#pragma unroll
        for (int ms = 0; ms < 2; ++ms) {
#pragma unroll
            for (int g = 0; g < 8; ++g) {
                const int row = m0 + ms * 16 + hl * 8 + g;
                if (row < M) {
                    float v = acc[ms][sub][g] + bias;
                    v = v > 0.0f ? v : 0.0f;
                    __bf16 h = (__bf16)v;
                    Chi[row * DHID + col] = h;
                    Clo[row * DHID + col] = (__bf16)(v - (float)h);
                }
            }
        }
    }
}

// ---------------------------------------------------------------------------
// Final layer: logits = root @ O_w^T + O_b, then log_softmax per batch row.
// M=8 -> WMMA cannot help; plain VALU. One block of 1024 threads per batch.
// ---------------------------------------------------------------------------
__global__ void __launch_bounds__(1024) logits_lsm_kernel(
    const __bf16* __restrict__ rhi, const __bf16* __restrict__ rlo,
    const float* __restrict__ Ow, const float* __restrict__ Ob,
    float* __restrict__ out)
{
    __shared__ float red[1024];
    const int b = blockIdx.x;
    const int o = threadIdx.x;
    const __bf16* rh = rhi + b * DHID;
    const __bf16* rl = rlo + b * DHID;
    const float*  w  = Ow + o * DHID;
    float acc = 0.0f;
    for (int k = 0; k < DHID; ++k) {
        float x = (float)rh[k] + (float)rl[k];
        acc = fmaf(x, w[k], acc);
    }
    const float logit = acc + Ob[o];

    red[o] = logit;
    __syncthreads();
    for (int s = 512; s > 0; s >>= 1) {
        if (o < s) red[o] = fmaxf(red[o], red[o + s]);
        __syncthreads();
    }
    const float mx = red[0];
    __syncthreads();
    red[o] = expf(logit - mx);
    __syncthreads();
    for (int s = 512; s > 0; s >>= 1) {
        if (o < s) red[o] += red[o + s];
        __syncthreads();
    }
    const float lse = mx + logf(red[0]);
    out[b * DHID + o] = logit - lse;
}

// ---------------------------------------------------------------------------
// Driver
// ---------------------------------------------------------------------------
extern "C" void kernel_launch(void* const* d_in, const int* in_sizes, int n_in,
                              void* d_out, int out_size, void* d_ws, size_t ws_size,
                              hipStream_t stream)
{
    (void)in_sizes; (void)n_in; (void)out_size; (void)ws_size;
    const float* leaves = (const float*)d_in[0];
    const float* V_w    = (const float*)d_in[1];
    const float* V_b    = (const float*)d_in[2];
    const float* U_w    = (const float*)d_in[3];
    const float* U_b    = (const float*)d_in[4];
    const float* W_w    = (const float*)d_in[5];
    const float* W_b    = (const float*)d_in[6];
    const float* O_w    = (const float*)d_in[7];
    const float* O_b    = (const float*)d_in[8];

    // Workspace carve-up (all 256B aligned):
    //   3 weight matrices hi/lo : 6 * 2MB = 12MB
    //   2 activation ping-pong buffers, each hi+lo : 4 * 16MB = 64MB
    char* ws = (char*)d_ws;
    size_t off = 0;
    auto carve = [&](size_t bytes) -> char* {
        char* p = ws + off;
        off += (bytes + 255) & ~(size_t)255;
        return p;
    };
    const size_t WB = (size_t)DHID * DHID * sizeof(__bf16);
    __bf16* Vhi = (__bf16*)carve(WB); __bf16* Vlo = (__bf16*)carve(WB);
    __bf16* Uhi = (__bf16*)carve(WB); __bf16* Ulo = (__bf16*)carve(WB);
    __bf16* Whi = (__bf16*)carve(WB); __bf16* Wlo = (__bf16*)carve(WB);
    const size_t HB = (size_t)NB * NL * DHID * sizeof(__bf16);
    __bf16* hAhi = (__bf16*)carve(HB); __bf16* hAlo = (__bf16*)carve(HB);
    __bf16* hBhi = (__bf16*)carve(HB); __bf16* hBlo = (__bf16*)carve(HB);

    // 1) Split weights to hi/lo bf16.
    {
        dim3 grid(DHID * DHID / 256);
        wsplit_kernel<<<grid, 256, 0, stream>>>(V_w, Vhi, Vlo, DHID * DHID);
        wsplit_kernel<<<grid, 256, 0, stream>>>(U_w, Uhi, Ulo, DHID * DHID);
        wsplit_kernel<<<grid, 256, 0, stream>>>(W_w, Whi, Wlo, DHID * DHID);
    }

    // 2) Leaf projection: h0 = relu(leaves @ V^T + V_b), M = 8192 rows.
    {
        const int M = NB * NL;
        dim3 grid(DHID / 64, (M + 255) / 256);
        tree_gemm<true, false><<<grid, 256, 0, stream>>>(
            leaves, nullptr, nullptr,
            Vhi, Vlo, V_b, nullptr, nullptr, nullptr,
            hAhi, hAlo, M, 1);
    }

    // 3) Tree reduction: 10 levels, M = 8*n for n = 512..1.
    __bf16* srcHi = hAhi; __bf16* srcLo = hAlo;
    __bf16* dstHi = hBhi; __bf16* dstLo = hBlo;
    for (int n = NL / 2; n >= 1; n >>= 1) {
        const int M = NB * n;
        dim3 grid(DHID / 64, (M + 255) / 256);
        tree_gemm<false, true><<<grid, 256, 0, stream>>>(
            nullptr, srcHi, srcLo,
            Uhi, Ulo, U_b, Whi, Wlo, W_b,
            dstHi, dstLo, M, 2);
        __bf16* t;
        t = srcHi; srcHi = dstHi; dstHi = t;
        t = srcLo; srcLo = dstLo; dstLo = t;
    }

    // 4) Output head + log_softmax. Roots are rows 0..7 of the final buffer.
    logits_lsm_kernel<<<NB, 1024, 0, stream>>>(srcHi, srcLo, O_w, O_b,
                                               (float*)d_out);
}
